// AffinityLoss_6828998001556
// MI455X (gfx1250) — compile-verified, matrix operand order checked by reading, not run on hardware
//
#include <hip/hip_runtime.h>

// AffinityLoss forward for MI455X (gfx1250, wave32, WMMA).
// loss = mean_b sum_{i,j} sgn(i)*sgn(j) * (X^T X)[i,j]^2,
// X[n,c] = (YY1[n]+eps)^(-1/4) * [V|Y][n,c],  sgn = +1 (V cols), -1 (Y cols).
//
// X is materialized FEATURE-MAJOR (X'[b][f][n], bf16) so the GEMM K-slab is a
// contiguous 16B-chunk copy -> GLOBAL_LOAD_ASYNC_TO_LDS_B128, double-buffered.

#define B_   8
#define N_   16384
#define D1_  256
#define D2_  64
#define DF_  320            // D1 + D2
#define EPSF 1e-12f

typedef __attribute__((ext_vector_type(16))) __bf16 v16bf;
typedef __attribute__((ext_vector_type(8)))  __bf16 v8bf;
typedef __attribute__((ext_vector_type(8)))  float  v8f;

#if defined(__has_builtin)
#if __has_builtin(__builtin_amdgcn_global_load_async_to_lds_b128) && \
    __has_builtin(__builtin_amdgcn_s_wait_asynccnt)
#define USE_ASYNC_LDS 1
#endif
#endif

#if defined(USE_ASYNC_LDS)
// Builtin signature (from hipcc diagnostic): params are pointers to
// 'int __attribute__((vector_size(16)))' in the global / LDS address spaces.
typedef int v4i_ __attribute__((vector_size(16)));
typedef __attribute__((address_space(1))) v4i_* g_v4i_p;
typedef __attribute__((address_space(3))) v4i_* l_v4i_p;
#define ASYNC_B128(gp, lp)                                                  \
    __builtin_amdgcn_global_load_async_to_lds_b128(                         \
        (g_v4i_p)(unsigned long long)(const void*)(gp),                     \
        (l_v4i_p)(unsigned int)(unsigned long long)(void*)(lp), 0, 0)
#endif

// ---------------------------------------------------------------------------
// Kernel 1: s[b,c] = sum_n Y[b,n,c]   (one block per batch, 256 threads)
// ---------------------------------------------------------------------------
__global__ void __launch_bounds__(256)
affinity_colsum(const float* __restrict__ Y, float* __restrict__ s)
{
    int b   = blockIdx.x;
    int t   = threadIdx.x;
    int col = t & 63;
    int rg  = t >> 6;                       // 4 row groups
    const float* Yb = Y + (size_t)b * N_ * D2_;
    float acc = 0.f;
    for (int n = rg; n < N_; n += 4)
        acc += Yb[(size_t)n * D2_ + col];
    __shared__ float red[4][64];
    red[rg][col] = acc;
    __syncthreads();
    if (t < 64)
        s[b * 64 + t] = red[0][t] + red[1][t] + red[2][t] + red[3][t];
}

// ---------------------------------------------------------------------------
// Kernel 2: X'[b][f][n] = bf16( (Y[n]·s+eps)^(-1/4) * [V|Y][n,f] ), transposed
// through LDS. One block = 32 rows of n; one wave owns one row at a time.
// Writes out 16B chunks along n (feature-major rows of length N).
// ---------------------------------------------------------------------------
__global__ void __launch_bounds__(256)
affinity_scale_cast_t(const float* __restrict__ V, const float* __restrict__ Y,
                      const float* __restrict__ s, __bf16* __restrict__ Xp)
{
    int b     = blockIdx.x >> 9;            // 512 blocks per batch
    int nbase = (blockIdx.x & 511) * 32;
    int t     = threadIdx.x;
    int lane  = t & 31;
    int wave  = t >> 5;

    __shared__ float ssh[64];
    __shared__ __align__(16) __bf16 T[DF_][40];   // 40-elem rows keep 16B align
    if (t < 64) ssh[t] = s[b * 64 + t];
    __syncthreads();

    #pragma unroll
    for (int r = 0; r < 4; ++r) {
        int nl = wave * 4 + r;
        int n  = nbase + nl;
        const float* Yrow = Y + ((size_t)b * N_ + n) * D2_;
        float2 y = *reinterpret_cast<const float2*>(Yrow + lane * 2);
        float p  = y.x * ssh[lane * 2] + y.y * ssh[lane * 2 + 1];
        #pragma unroll
        for (int o = 16; o > 0; o >>= 1) p += __shfl_xor(p, o, 32);
        float sc = rsqrtf(sqrtf(p + EPSF));         // (YY1+eps)^(-1/4)

        const float* Vrow = V + ((size_t)b * N_ + n) * D1_;
        float4 v0 = *reinterpret_cast<const float4*>(Vrow + lane * 8);
        float4 v1 = *reinterpret_cast<const float4*>(Vrow + lane * 8 + 4);
        T[lane * 8 + 0][nl] = (__bf16)(v0.x * sc);
        T[lane * 8 + 1][nl] = (__bf16)(v0.y * sc);
        T[lane * 8 + 2][nl] = (__bf16)(v0.z * sc);
        T[lane * 8 + 3][nl] = (__bf16)(v0.w * sc);
        T[lane * 8 + 4][nl] = (__bf16)(v1.x * sc);
        T[lane * 8 + 5][nl] = (__bf16)(v1.y * sc);
        T[lane * 8 + 6][nl] = (__bf16)(v1.z * sc);
        T[lane * 8 + 7][nl] = (__bf16)(v1.w * sc);
        T[D1_ + lane * 2 + 0][nl] = (__bf16)(y.x * sc);
        T[D1_ + lane * 2 + 1][nl] = (__bf16)(y.y * sc);
    }
    __syncthreads();

    // 1280 16B chunks: chunk c -> feature f = c/4, n0 = (c%4)*8
    #pragma unroll
    for (int i = 0; i < 5; ++i) {
        int c  = t + 256 * i;
        int f  = c >> 2;
        int n0 = (c & 3) * 8;
        v8bf val = *reinterpret_cast<const v8bf*>(&T[f][n0]);
        *reinterpret_cast<v8bf*>(Xp + ((size_t)b * DF_ + f) * N_ + nbase + n0) = val;
    }
}

// ---------------------------------------------------------------------------
// Kernel 3: batched 320x320 Gram M = X^T X via v_wmma_f32_16x16x32_bf16 with
// fused signed-square reduction. Grid (5,5,B): 64x64 output block / WG,
// 8 waves: wave w -> tile row (w&3), tile cols 2*(w>>2)..+1.
// K-slabs (32 deep) are straight 16B copies from feature-major X' into LDS;
// double-buffered GLOBAL_LOAD_ASYNC_TO_LDS_B128 when available.
// Fragments: two ds_load_b128 per matrix per the documented 16-bit layout
// (lane<16: K {0..7,16..23}; lane>=16: K {8..15,24..31}).
// ---------------------------------------------------------------------------
__global__ void __launch_bounds__(256)
affinity_gram_loss(const __bf16* __restrict__ Xp, float* __restrict__ partial)
{
    const int b     = blockIdx.z;
    const int Mbase = blockIdx.x * 64;
    const int Nbase = blockIdx.y * 64;
    const int t     = threadIdx.x;
    const int lane  = t & 31;
    const int wave  = t >> 5;
    const int tr    = wave & 3;
    const int tc0   = (wave >> 2) * 2;

    __shared__ __align__(16) __bf16 smA[2][64][40];   // 80B rows: 16B-aligned octets
    __shared__ __align__(16) __bf16 smB[2][64][40];
    __shared__ float wsum[8];

    // cooperative copy mapping: thread t -> feature (t>>2), k-octet (t&3)*8
    const int fl   = t >> 2;
    const int koff = (t & 3) * 8;
    const __bf16* gA = Xp + ((size_t)b * DF_ + Mbase + fl) * N_ + koff;
    const __bf16* gB = Xp + ((size_t)b * DF_ + Nbase + fl) * N_ + koff;

    const int r  = lane & 15;
    const int kh = lane >> 4;

    v8f acc0 = {}, acc1 = {};

    auto compute = [&](int bi) {
        v8bf a0 = *reinterpret_cast<const v8bf*>(&smA[bi][tr * 16 + r][kh * 8]);
        v8bf a1 = *reinterpret_cast<const v8bf*>(&smA[bi][tr * 16 + r][16 + kh * 8]);
        v16bf af = __builtin_shufflevector(a0, a1,
            0, 1, 2, 3, 4, 5, 6, 7, 8, 9, 10, 11, 12, 13, 14, 15);

        v8bf b00 = *reinterpret_cast<const v8bf*>(&smB[bi][tc0 * 16 + r][kh * 8]);
        v8bf b01 = *reinterpret_cast<const v8bf*>(&smB[bi][tc0 * 16 + r][16 + kh * 8]);
        v16bf bf0 = __builtin_shufflevector(b00, b01,
            0, 1, 2, 3, 4, 5, 6, 7, 8, 9, 10, 11, 12, 13, 14, 15);

        v8bf b10 = *reinterpret_cast<const v8bf*>(&smB[bi][(tc0 + 1) * 16 + r][kh * 8]);
        v8bf b11 = *reinterpret_cast<const v8bf*>(&smB[bi][(tc0 + 1) * 16 + r][16 + kh * 8]);
        v16bf bf1 = __builtin_shufflevector(b10, b11,
            0, 1, 2, 3, 4, 5, 6, 7, 8, 9, 10, 11, 12, 13, 14, 15);

        acc0 = __builtin_amdgcn_wmma_f32_16x16x32_bf16(
                   false, af, false, bf0, (short)0, acc0, false, false);
        acc1 = __builtin_amdgcn_wmma_f32_16x16x32_bf16(
                   false, af, false, bf1, (short)0, acc1, false, false);
    };

    const int S = N_ / 32;                     // 512 K-steps

#if defined(USE_ASYNC_LDS)
    // Double-buffered async pipeline: issue slab i+1, wait slab i, compute.
    ASYNC_B128(gA, &smA[0][fl][koff]);
    ASYNC_B128(gB, &smB[0][fl][koff]);
    int buf = 0;
    for (int i = 0; i < S - 1; ++i) {
        ASYNC_B128(gA + (size_t)(i + 1) * 32, &smA[buf ^ 1][fl][koff]);
        ASYNC_B128(gB + (size_t)(i + 1) * 32, &smB[buf ^ 1][fl][koff]);
        __builtin_amdgcn_s_wait_asynccnt(2);   // slab i landed; i+1 in flight
        __syncthreads();
        compute(buf);
        __syncthreads();                       // all reads done before reuse
        buf ^= 1;
    }
    __builtin_amdgcn_s_wait_asynccnt(0);
    __syncthreads();
    compute(buf);
#else
    // Fallback: synchronous vector copy through VGPRs (still all-b128).
    for (int i = 0; i < S; ++i) {
        v8bf ga = *reinterpret_cast<const v8bf*>(gA + (size_t)i * 32);
        v8bf gb = *reinterpret_cast<const v8bf*>(gB + (size_t)i * 32);
        __syncthreads();
        *reinterpret_cast<v8bf*>(&smA[0][fl][koff]) = ga;
        *reinterpret_cast<v8bf*>(&smB[0][fl][koff]) = gb;
        __syncthreads();
        compute(0);
    }
#endif

    // Epilogue. C/D layout: lane l -> N = l%16, VGPR v -> M = v + 8*(l/16).
    const int iM  = Mbase + tr * 16 + 8 * kh;
    const int jN0 = Nbase + tc0 * 16 + (lane & 15);
    const int jN1 = jN0 + 16;
    const float sj0 = (jN0 < D1_) ? 1.f : -1.f;
    const float sj1 = (jN1 < D1_) ? 1.f : -1.f;
    float sum = 0.f;
    #pragma unroll
    for (int v = 0; v < 8; ++v) {
        float si = ((iM + v) < D1_) ? 1.f : -1.f;
        sum += si * (sj0 * acc0[v] * acc0[v] + sj1 * acc1[v] * acc1[v]);
    }
    #pragma unroll
    for (int o = 16; o > 0; o >>= 1) sum += __shfl_xor(sum, o, 32);

    if (lane == 0) wsum[wave] = sum;
    __syncthreads();
    if (t == 0) {
        float tot = 0.f;
        #pragma unroll
        for (int w = 0; w < 8; ++w) tot += wsum[w];
        partial[(blockIdx.z * 5 + blockIdx.y) * 5 + blockIdx.x] = tot;
    }
}

// ---------------------------------------------------------------------------
// Kernel 4: deterministic fixed-order reduction of 200 partials -> loss.
// ---------------------------------------------------------------------------
__global__ void __launch_bounds__(256)
affinity_reduce(const float* __restrict__ partial, float* __restrict__ out)
{
    __shared__ float sh[256];
    int t = threadIdx.x;
    sh[t] = (t < 200) ? partial[t] : 0.f;
    __syncthreads();
    #pragma unroll
    for (int o = 128; o > 0; o >>= 1) {
        if (t < o) sh[t] += sh[t + o];
        __syncthreads();
    }
    if (t == 0) out[0] = sh[0] * (1.0f / B_);
}

// ---------------------------------------------------------------------------
extern "C" void kernel_launch(void* const* d_in, const int* in_sizes, int n_in,
                              void* d_out, int out_size, void* d_ws, size_t ws_size,
                              hipStream_t stream)
{
    const float* V = (const float*)d_in[0];   // (8, 16384, 256) f32
    const float* Y = (const float*)d_in[1];   // (8, 16384, 64)  f32
    float* out = (float*)d_out;               // scalar loss

    char*   ws      = (char*)d_ws;
    __bf16* Xbuf    = (__bf16*)ws;                                  // 8*320*16384*2 = 80 MiB
    float*  sbuf    = (float*)(ws + (size_t)B_ * N_ * DF_ * 2);     // 2 KiB
    float*  partial = sbuf + B_ * 64;                               // 800 B

    affinity_colsum      <<<B_,             256, 0, stream>>>(Y, sbuf);
    affinity_scale_cast_t<<<B_ * (N_ / 32), 256, 0, stream>>>(V, Y, sbuf, Xbuf);
    affinity_gram_loss   <<<dim3(5, 5, B_), 256, 0, stream>>>(Xbuf, partial);
    affinity_reduce      <<<1,              256, 0, stream>>>(partial, out);
}
